// AdvancedQuantumFeatureMap_21869973471235
// MI455X (gfx1250) — matrix operand order repaired and checked by reading, not compiled, not run on hardware
//
#include <hip/hip_runtime.h>
#include <hip/hip_bf16.h>
#include <math.h>

#define B_ROWS 16384
#define QDIM   1024
#define FDIM   2
#define DDIM   5

typedef __attribute__((ext_vector_type(16))) __bf16 v16bf;
typedef __attribute__((ext_vector_type(8)))  __bf16 v8bf;
typedef __attribute__((ext_vector_type(8)))  float  v8f;
typedef __attribute__((ext_vector_type(4)))  int    v4i;

typedef __attribute__((address_space(1))) v4i gv4i;
typedef __attribute__((address_space(3))) v4i lv4i;

#if defined(__has_builtin)
# if __has_builtin(__builtin_amdgcn_global_load_async_to_lds_b128) && \
     __has_builtin(__builtin_amdgcn_s_wait_asynccnt)
#  define USE_ASYNC_COPY 1
# endif
#endif
#ifndef USE_ASYNC_COPY
# define USE_ASYNC_COPY 0
#endif

__device__ __forceinline__ unsigned short f2bf(float f) {
  unsigned int u = __builtin_bit_cast(unsigned int, f);
  u += 0x7FFFu + ((u >> 16) & 1u);
  return (unsigned short)(u >> 16);
}

#if USE_ASYNC_COPY
// async DMA: 16 bytes global -> LDS (tracked by ASYNCcnt)
__device__ __forceinline__ void async_cp16(const unsigned short* g, unsigned short* l) {
  __builtin_amdgcn_global_load_async_to_lds_b128(
      (gv4i*)(void*)g, (lv4i*)(void*)l, 0, 0);
}
#endif

// ------------------------------------------- f32 [N,K] -> bf16 transposed [K,N]
__global__ void cvt_t_bf16_kernel(const float* __restrict__ src,
                                  unsigned short* __restrict__ dst,
                                  int N, int K) {
  int i = blockIdx.x * blockDim.x + threadIdx.x;
  int stride = gridDim.x * blockDim.x;
  int total = N * K;
  for (; i < total; i += stride) {
    int n = i / K, k = i - n * K;           // read coalesced along K
    dst[(size_t)k * N + n] = f2bf(src[i]);
  }
}

// ------------------------------------------------- layer1: matvec(K=2)+LN+act
// act: 0 = exact GELU, 1 = SiLU.  out bf16 [B,H], one block per row.
__global__ __launch_bounds__(256)
void layer1_kernel(const float* __restrict__ x,   // [B,2]
                   const float* __restrict__ W,   // [H,2]
                   const float* __restrict__ b,
                   const float* __restrict__ g,
                   const float* __restrict__ be,
                   unsigned short* __restrict__ out,
                   int H, int act) {
  const int row = blockIdx.x;
  const int tid = threadIdx.x;
  const int per = H >> 8;                       // 16 or 8
  const float x0 = x[row * FDIM + 0];
  const float x1 = x[row * FDIM + 1];

  float pre[16];
  float s = 0.f, ss = 0.f;
  for (int i = 0; i < per; ++i) {
    int j = tid + (i << 8);
    float p = fmaf(x0, W[j * 2 + 0], fmaf(x1, W[j * 2 + 1], b[j]));
    pre[i] = p; s += p; ss += p * p;
  }
  for (int o = 16; o > 0; o >>= 1) { s += __shfl_down(s, o); ss += __shfl_down(ss, o); }
  __shared__ float rs[8], rss[8], s_m, s_inv;
  const int wid = tid >> 5, lane = tid & 31;
  if (lane == 0) { rs[wid] = s; rss[wid] = ss; }
  __syncthreads();
  if (tid == 0) {
    float S = 0.f, SS = 0.f;
    for (int i = 0; i < 8; ++i) { S += rs[i]; SS += rss[i]; }
    float m = S / (float)H;
    float v = SS / (float)H - m * m;
    s_m = m; s_inv = rsqrtf(v + 1e-5f);
  }
  __syncthreads();
  const float m = s_m, inv = s_inv;
  for (int i = 0; i < per; ++i) {
    int j = tid + (i << 8);
    float y = (pre[i] - m) * inv * g[j] + be[j];
    float a = (act == 0) ? 0.5f * y * (1.f + erff(y * 0.70710678118654752f))
                         : y / (1.f + expf(-y));
    out[(size_t)row * H + j] = f2bf(a);
  }
}

// ----------------------------------------------- LN + tanh (h2), f32 -> bf16
__global__ __launch_bounds__(256)
void ln_tanh_kernel(const float* __restrict__ in,   // [B,H]
                    const float* __restrict__ g,
                    const float* __restrict__ be,
                    unsigned short* __restrict__ out, int H) {
  const int row = blockIdx.x;
  const int tid = threadIdx.x;
  const int per = H >> 8;
  const float* prow = in + (size_t)row * H;

  float pre[16];
  float s = 0.f, ss = 0.f;
  for (int i = 0; i < per; ++i) {
    int j = tid + (i << 8);
    float p = prow[j];
    pre[i] = p; s += p; ss += p * p;
  }
  for (int o = 16; o > 0; o >>= 1) { s += __shfl_down(s, o); ss += __shfl_down(ss, o); }
  __shared__ float rs[8], rss[8], s_m, s_inv;
  const int wid = tid >> 5, lane = tid & 31;
  if (lane == 0) { rs[wid] = s; rss[wid] = ss; }
  __syncthreads();
  if (tid == 0) {
    float S = 0.f, SS = 0.f;
    for (int i = 0; i < 8; ++i) { S += rs[i]; SS += rss[i]; }
    float m = S / (float)H;
    float v = SS / (float)H - m * m;
    s_m = m; s_inv = rsqrtf(v + 1e-5f);
  }
  __syncthreads();
  const float m = s_m, inv = s_inv;
  for (int i = 0; i < per; ++i) {
    int j = tid + (i << 8);
    float y = (pre[i] - m) * inv * g[j] + be[j];
    out[(size_t)row * H + j] = f2bf(tanhf(y));
  }
}

// --------------------------------------------------- qs = (sin+cos+tanh)/3
__global__ void qs_kernel(const float* __restrict__ amp,
                          const float* __restrict__ phase,   // already tanh'd
                          const float* __restrict__ rot_freq,
                          const float* __restrict__ rot_phase,
                          unsigned short* __restrict__ out, int total) {
  int i = blockIdx.x * blockDim.x + threadIdx.x;
  if (i >= total) return;
  int q = i & (QDIM - 1);
  int base = (DDIM - 1) * QDIM * 3 + q * 3;
  float rx = sinf(fmaf(amp[i],   rot_freq[base + 0], rot_phase[base + 0]));
  float ry = cosf(fmaf(phase[i], rot_freq[base + 1], rot_phase[base + 1]));
  float rz = tanhf(rot_phase[base + 2]);
  out[i] = f2bf((rx + ry + rz) * (1.f / 3.f));
}

// ------------------------------------------------------------- WMMA GEMM
// C[M,N] = epi(A[M,K] @ Wt[K,N] + bias)
// A bf16 row-major [M,K]; Wt bf16 row-major [K,N] (pre-transposed weights).
// EPI: 0 -> f32 store, 1 -> tanh + f32 store, 2 -> bf16 store.
template <int EPI>
__global__ __launch_bounds__(256)
void gemm_bf16_kernel(const unsigned short* __restrict__ A,
                      const unsigned short* __restrict__ Wt,
                      const float* __restrict__ bias,
                      float* __restrict__ outF,
                      unsigned short* __restrict__ outH,
                      int M, int N, int K) {
  constexpr int BM = 128, BN = 128, BK = 64;
  constexpr int LDA = BK + 8;   // 72 ushorts = 144B (16B aligned rows)
  constexpr int LDB = BN + 8;   // 136 ushorts = 272B (16B aligned rows)
  __shared__ __align__(16) unsigned short lA[2][BM * LDA];
  __shared__ __align__(16) unsigned short lB[2][BK * LDB];

  const int tid  = threadIdx.x;
  const int lane = tid & 31;
  const int wave = tid >> 5;
  const int wr = wave & 1;      // wave row  (M: 2 x 64)
  const int wc = wave >> 1;     // wave col  (N: 4 x 32)
  const int bm = blockIdx.y * BM;
  const int bn = blockIdx.x * BN;

  v8f acc[4][2];
#pragma unroll
  for (int i = 0; i < 4; ++i)
#pragma unroll
    for (int j = 0; j < 2; ++j) acc[i][j] = {};

  // staging geometry: A 128x64 (2 thr/row x 32 el), B 64x128 (4 thr/row x 32 el)
  const int arow_g = tid >> 1;
  const int acol_g = (tid & 1) << 5;      // 0 / 32
  const int brow_g = tid >> 2;
  const int bcol_g = (tid & 3) << 5;      // 0 / 32 / 64 / 96

  const unsigned short* Ag = &A[(size_t)(bm + arow_g) * K + acol_g];
  const unsigned short* Bg = &Wt[(size_t)brow_g * N + bn + bcol_g];

  const int arow  = lane & 15;
  const int khalf = (lane >> 4) << 3;     // 0 or 8
  const int nk = K / BK;

#if USE_ASYNC_COPY
  // ---- async DMA global -> LDS (ASYNCcnt), double buffered -------------
  {
    unsigned short* la = &lA[0][arow_g * LDA + acol_g];
    unsigned short* lb = &lB[0][brow_g * LDB + bcol_g];
#pragma unroll
    for (int c = 0; c < 4; ++c) async_cp16(Ag + c * 8, la + c * 8);
#pragma unroll
    for (int c = 0; c < 4; ++c) async_cp16(Bg + c * 8, lb + c * 8);
  }
  for (int t = 0; t < nk; ++t) {
    const int buf = t & 1;
    __builtin_amdgcn_s_wait_asynccnt(0);
    __syncthreads();
    if (t + 1 < nk) {
      const int nb = (t + 1) & 1;
      const unsigned short* a = Ag + (t + 1) * BK;
      const unsigned short* b = Bg + (size_t)(t + 1) * BK * N;
      unsigned short* la = &lA[nb][arow_g * LDA + acol_g];
      unsigned short* lb = &lB[nb][brow_g * LDB + bcol_g];
#pragma unroll
      for (int c = 0; c < 4; ++c) async_cp16(a + c * 8, la + c * 8);
#pragma unroll
      for (int c = 0; c < 4; ++c) async_cp16(b + c * 8, lb + c * 8);
    }
#else
  // ---- register-staged double buffering --------------------------------
  uint4 ra[4], rb[4];
  {
#pragma unroll
    for (int c = 0; c < 4; ++c) ra[c] = *(const uint4*)(Ag + c * 8);
#pragma unroll
    for (int c = 0; c < 4; ++c) rb[c] = *(const uint4*)(Bg + c * 8);
  }
  for (int t = 0; t < nk; ++t) {
    const int buf = t & 1;
    {
      unsigned short* la = &lA[buf][arow_g * LDA + acol_g];
      unsigned short* lb = &lB[buf][brow_g * LDB + bcol_g];
#pragma unroll
      for (int c = 0; c < 4; ++c) *(uint4*)(la + c * 8) = ra[c];
#pragma unroll
      for (int c = 0; c < 4; ++c) *(uint4*)(lb + c * 8) = rb[c];
    }
    __syncthreads();
    if (t + 1 < nk) {
      const unsigned short* a = Ag + (t + 1) * BK;
      const unsigned short* b = Bg + (size_t)(t + 1) * BK * N;
#pragma unroll
      for (int c = 0; c < 4; ++c) ra[c] = *(const uint4*)(a + c * 8);
#pragma unroll
      for (int c = 0; c < 4; ++c) rb[c] = *(const uint4*)(b + c * 8);
    }
#endif
    // ---- 16 WMMAs on buffer `buf` (two K=32 sub-steps) -----------------
#pragma unroll
    for (int ks = 0; ks < BK; ks += 32) {
      // A fragments: lanes 0-15 -> K ks+0..7 & ks+16..23 ; lanes 16-31 -> +8
      v16bf afr[4];
#pragma unroll
      for (int mi = 0; mi < 4; ++mi) {
        const unsigned short* p = &lA[buf][(wr * 64 + mi * 16 + arow) * LDA + ks + khalf];
        v8bf lo = *(const v8bf*)(p);
        v8bf hi = *(const v8bf*)(p + 16);
        afr[mi] = __builtin_shufflevector(lo, hi, 0,1,2,3,4,5,6,7,8,9,10,11,12,13,14,15);
      }
      // B fragments: N = lane&15; lanes 0-15 -> K ks+0..7 & ks+16..23 ; lanes 16-31 -> +8
      v16bf bfr[2];
#pragma unroll
      for (int ni = 0; ni < 2; ++ni) {
        const unsigned short* p = &lB[buf][(ks + khalf) * LDB + wc * 32 + ni * 16 + arow];
#pragma unroll
        for (int s = 0; s < 16; ++s)
          bfr[ni][s] = __builtin_bit_cast(__bf16, p[(s < 8 ? s : s + 8) * LDB]);
      }
#pragma unroll
      for (int mi = 0; mi < 4; ++mi)
#pragma unroll
        for (int ni = 0; ni < 2; ++ni)
          acc[mi][ni] = __builtin_amdgcn_wmma_f32_16x16x32_bf16(
              false, afr[mi], false, bfr[ni], (short)0, acc[mi][ni], false, false);
    }
    __syncthreads();
  }

  // epilogue: VGPR r -> M = r (lanes 0-15) or r+8 (lanes 16-31); N = lane&15
  const int crow = (lane < 16) ? 0 : 8;
  const int ccol = lane & 15;
#pragma unroll
  for (int mi = 0; mi < 4; ++mi) {
#pragma unroll
    for (int ni = 0; ni < 2; ++ni) {
      const int nidx = bn + wc * 32 + ni * 16 + ccol;
      const float bv = bias[nidx];
#pragma unroll
      for (int r = 0; r < 8; ++r) {
        const int midx = bm + wr * 64 + mi * 16 + crow + r;
        float v = acc[mi][ni][r] + bv;
        if (EPI == 1) v = tanhf(v);
        if (EPI == 2) outH[(size_t)midx * N + nidx] = f2bf(v);
        else          outF[(size_t)midx * N + nidx] = v;
      }
    }
  }
}

// ---------------------------------------------------------------- launcher
extern "C" void kernel_launch(void* const* d_in, const int* in_sizes, int n_in,
                              void* d_out, int out_size, void* d_ws, size_t ws_size,
                              hipStream_t stream) {
  (void)in_sizes; (void)n_in; (void)out_size; (void)ws_size;
  const float* x         = (const float*)d_in[0];
  const float* amp_W1    = (const float*)d_in[1];
  const float* amp_b1    = (const float*)d_in[2];
  const float* amp_g1    = (const float*)d_in[3];
  const float* amp_be1   = (const float*)d_in[4];
  const float* amp_W2    = (const float*)d_in[5];
  const float* amp_b2    = (const float*)d_in[6];
  const float* amp_g2    = (const float*)d_in[7];
  const float* amp_be2   = (const float*)d_in[8];
  const float* amp_W3    = (const float*)d_in[9];
  const float* amp_b3    = (const float*)d_in[10];
  const float* ph_W1     = (const float*)d_in[11];
  const float* ph_b1     = (const float*)d_in[12];
  const float* ph_g1     = (const float*)d_in[13];
  const float* ph_be1    = (const float*)d_in[14];
  const float* ph_W2     = (const float*)d_in[15];
  const float* ph_b2     = (const float*)d_in[16];
  const float* rot_freq  = (const float*)d_in[17];
  const float* rot_phase = (const float*)d_in[18];
  const float* attn_in_w = (const float*)d_in[19];
  const float* attn_in_b = (const float*)d_in[20];
  const float* attn_out_w= (const float*)d_in[21];
  const float* attn_out_b= (const float*)d_in[22];

  char* ws = (char*)d_ws;
  // pre-transposed bf16 weight copies  Wt[K,N]
  unsigned short* W2t    = (unsigned short*)(ws + 0);           // 16 MB [4096,2048]
  unsigned short* W3t    = (unsigned short*)(ws + 16777216);    // 4 MB  [2048,1024]
  unsigned short* phW2t  = (unsigned short*)(ws + 20971520);    // 4 MB  [2048,1024]
  unsigned short* attnVt = (unsigned short*)(ws + 25165824);    // 2 MB  [1024,1024]
  unsigned short* attnOt = (unsigned short*)(ws + 27262976);    // 2 MB  [1024,1024]
  // activations (with reuse)
  unsigned short* h1h    = (unsigned short*)(ws + 29360128);            // 128 MB bf16 [B,4096]
  unsigned short* p1h    = (unsigned short*)(ws + 163577856);           // 64 MB  bf16 [B,2048]
  float*          h2raw  = (float*)(ws + 230686720);                    // 128 MB f32  [B,2048]
  unsigned short* h2h    = (unsigned short*)(ws + 29360128);            // reuse h1 region
  float*          ampf   = (float*)(ws + 230686720);                    // reuse h2raw region
  float*          phasef = (float*)(ws + 230686720 + 67108864);
  unsigned short* qsh    = (unsigned short*)(ws + 163577856);           // reuse p1 region
  unsigned short* vh     = (unsigned short*)(ws + 163577856 + 33554432);
  float*          outF   = (float*)d_out;

  const int M = B_ROWS;

  // ---- weight conversions, f32 [N,K] -> bf16 [K,N]
  cvt_t_bf16_kernel<<<4096, 256, 0, stream>>>(amp_W2,  W2t,   2048, 4096);
  cvt_t_bf16_kernel<<<2048, 256, 0, stream>>>(amp_W3,  W3t,   1024, 2048);
  cvt_t_bf16_kernel<<<2048, 256, 0, stream>>>(ph_W2,   phW2t, 1024, 2048);
  cvt_t_bf16_kernel<<<1024, 256, 0, stream>>>(attn_in_w + 2 * QDIM * QDIM, attnVt, QDIM, QDIM);
  cvt_t_bf16_kernel<<<1024, 256, 0, stream>>>(attn_out_w, attnOt, QDIM, QDIM);

  // ---- layer 1 (K=2 matvec + LN + act), fused
  layer1_kernel<<<M, 256, 0, stream>>>(x, amp_W1, amp_b1, amp_g1, amp_be1, h1h, 4096, /*gelu*/0);
  layer1_kernel<<<M, 256, 0, stream>>>(x, ph_W1,  ph_b1,  ph_g1,  ph_be1,  p1h, 2048, /*silu*/1);

  // ---- h2raw = h1 @ W2^T + b2   (16384 x 2048 x 4096)
  gemm_bf16_kernel<0><<<dim3(2048 / 128, M / 128), 256, 0, stream>>>(
      h1h, W2t, amp_b2, h2raw, nullptr, M, 2048, 4096);
  // ---- h2 = tanh(LN(h2raw))
  ln_tanh_kernel<<<M, 256, 0, stream>>>(h2raw, amp_g2, amp_be2, h2h, 2048);

  // ---- amp = h2 @ W3^T + b3    (16384 x 1024 x 2048)
  gemm_bf16_kernel<0><<<dim3(1024 / 128, M / 128), 256, 0, stream>>>(
      h2h, W3t, amp_b3, ampf, nullptr, M, 1024, 2048);

  // ---- phase = tanh(p1 @ phW2^T + pb2)   (16384 x 1024 x 2048)
  gemm_bf16_kernel<1><<<dim3(1024 / 128, M / 128), 256, 0, stream>>>(
      p1h, phW2t, ph_b2, phasef, nullptr, M, 1024, 2048);

  // ---- qs = (sin + cos + tanh)/3 -> bf16
  qs_kernel<<<(M * QDIM) / 256, 256, 0, stream>>>(ampf, phasef, rot_freq, rot_phase,
                                                  qsh, M * QDIM);

  // ---- v = qs @ attnV^T + bV (only rows [2Q:3Q) of attn_in_w matter)
  gemm_bf16_kernel<2><<<dim3(1024 / 128, M / 128), 256, 0, stream>>>(
      qsh, attnVt, attn_in_b + 2 * QDIM, nullptr, vh, M, 1024, 1024);

  // ---- out = v @ attnOut^T + bOut  -> f32 d_out
  gemm_bf16_kernel<0><<<dim3(1024 / 128, M / 128), 256, 0, stream>>>(
      vh, attnOt, attn_out_b, outF, nullptr, M, 1024, 1024);
}
